// RoPEMaskedAttentionHead_4690104287852
// MI455X (gfx1250) — matrix operand order repaired and checked by reading, not compiled, not run on hardware
//
#include <hip/hip_runtime.h>
#include <hip/hip_bf16.h>

typedef __attribute__((ext_vector_type(16))) _Float16 v16h;
typedef __attribute__((ext_vector_type(8)))  _Float16 v8h;
typedef __attribute__((ext_vector_type(8)))  float    v8f;
typedef __attribute__((ext_vector_type(4)))  float    f32x4;
typedef unsigned int u32x4 __attribute__((ext_vector_type(4)));
typedef int          i32x4 __attribute__((ext_vector_type(4)));
typedef int          i32x8 __attribute__((ext_vector_type(8)));

#define SEQ_LEN   4096
#define D_MODEL   2048
#define HEAD_DIM  128
#define BATCH     4

#if defined(__has_builtin)
#if __has_builtin(__builtin_amdgcn_tensor_load_to_lds) && \
    __has_builtin(__builtin_amdgcn_s_wait_tensorcnt)
#define HAVE_TDM 1
#endif
#endif
#ifndef HAVE_TDM
#define HAVE_TDM 0
#endif

__device__ __forceinline__ v8f wmma_f16(v16h a, v16h b, v8f c) {
  // D = A(16x32 f16) * B(32x16 f16) + C(16x16 f32)
  return __builtin_amdgcn_wmma_f32_16x16x32_f16(false, a, false, b, (short)0, c,
                                                false, false);
}

#if HAVE_TDM
// Issue one TDM load of a 2D tile (f16 elements) from global into LDS.
// ISA ch. 8: D# group0 = {count|flags, lds_addr, global_addr, type=2},
// group1 = {mask/data_size, tensor_dim0/1, tile_dim0/1/2, dim strides}.
__device__ __forceinline__ void tdm_load_2d(unsigned lds_addr, const void* gptr,
                                            unsigned tile_d0, unsigned tile_d1,
                                            unsigned tensor_d0, unsigned tensor_d1,
                                            unsigned long stride0_elems) {
  unsigned long ga = (unsigned long)gptr;
  u32x4 g0;
  g0[0] = 1u;                                             // count=1, user D#
  g0[1] = lds_addr;                                       // LDS byte address
  g0[2] = (unsigned)(ga & 0xffffffffu);                   // global_addr[31:0]
  g0[3] = (unsigned)((ga >> 32) & 0x01ffffffu) | (2u << 30);  // [56:32] | type=2
  i32x8 g1;
  g1[0] = (int)(1u << 16);                                // data_size=1 -> 2B
  g1[1] = (int)((tensor_d0 & 0xffffu) << 16);             // tensor_dim0[15:0]
  g1[2] = (int)((tensor_d0 >> 16) | ((tensor_d1 & 0xffffu) << 16));
  g1[3] = (int)((tensor_d1 >> 16) | ((tile_d0 & 0xffffu) << 16));
  g1[4] = (int)(tile_d1 & 0xffffu);                       // tile_dim1, dim2=0
  g1[5] = (int)(stride0_elems & 0xffffffffu);             // dim0 stride lo
  g1[6] = (int)((stride0_elems >> 32) & 0xffffu);         // dim0 stride hi
  g1[7] = 0;                                              // dim1 stride (2D)
  i32x4 z4 = {0, 0, 0, 0};
#if __has_include(<hip/amd_detail/amd_gfx1250_TDM.h>)
  i32x8 z8 = {0, 0, 0, 0, 0, 0, 0, 0};
  __builtin_amdgcn_tensor_load_to_lds(g0, g1, z4, z4, z8, 0);
#else
  __builtin_amdgcn_tensor_load_to_lds(g0, g1, z4, z4, 0);
#endif
}
#endif  // HAVE_TDM

// ---------------------------------------------------------------------------
// Kernel 1: transpose f32 weights (128 x 2048) -> f16 wT (2048 x 128)
// ---------------------------------------------------------------------------
__global__ void __launch_bounds__(256) wtranspose_kernel(
    const float* __restrict__ wq, const float* __restrict__ wk,
    const float* __restrict__ wv,
    _Float16* __restrict__ wqT, _Float16* __restrict__ wkT,
    _Float16* __restrict__ wvT) {
  const float* w = (blockIdx.y == 0) ? wq : (blockIdx.y == 1) ? wk : wv;
  _Float16*    o = (blockIdx.y == 0) ? wqT : (blockIdx.y == 1) ? wkT : wvT;
  int idx = blockIdx.x * 256 + threadIdx.x;
  int d = idx >> 7;
  int h = idx & 127;
  o[(size_t)d * HEAD_DIM + h] = (_Float16)w[(size_t)h * D_MODEL + d];
}

// ---------------------------------------------------------------------------
// Kernel 2: fused QKV projection (WMMA) + RoPE.
// ---------------------------------------------------------------------------
__global__ void __launch_bounds__(256) proj_rope_kernel(
    const float* __restrict__ x,
    const _Float16* __restrict__ wqT, const _Float16* __restrict__ wkT,
    const _Float16* __restrict__ wvT,
    _Float16* __restrict__ qg, _Float16* __restrict__ kTg,
    _Float16* __restrict__ vg) {
  __shared__ __align__(32) float qs[16][HEAD_DIM];
  __shared__ __align__(32) float ks[16][HEAD_DIM];

  const int tid  = threadIdx.x;
  const int wave = tid >> 5;
  const int lane = tid & 31;
  const int n0   = wave * 16;
  const int mrow = lane & 15;
  const int koff = (lane >> 4) * 8;

  const long r0 = (long)blockIdx.x * 16;
  const float* xrow = x + (r0 + mrow) * (long)D_MODEL;

  v8f accq = {}, acck = {}, accv = {};

  for (int kb = 0; kb < D_MODEL; kb += 32) {
    f32x4 x0 = *(const f32x4*)(xrow + kb + koff);
    f32x4 x1 = *(const f32x4*)(xrow + kb + koff + 4);
    f32x4 x2 = *(const f32x4*)(xrow + kb + 16 + koff);
    f32x4 x3 = *(const f32x4*)(xrow + kb + 16 + koff + 4);
    v16h a;
#pragma unroll
    for (int i = 0; i < 4; ++i) {
      a[i]      = (_Float16)x0[i];
      a[4 + i]  = (_Float16)x1[i];
      a[8 + i]  = (_Float16)x2[i];
      a[12 + i] = (_Float16)x3[i];
    }
    v16h bq, bk, bv;
#pragma unroll
    for (int i = 0; i < 8; ++i) {
      const size_t o0 = (size_t)(kb + koff + i) * HEAD_DIM + n0 + mrow;
      const size_t o1 = (size_t)(kb + 16 + koff + i) * HEAD_DIM + n0 + mrow;
      bq[i] = wqT[o0]; bq[8 + i] = wqT[o1];
      bk[i] = wkT[o0]; bk[8 + i] = wkT[o1];
      bv[i] = wvT[o0]; bv[8 + i] = wvT[o1];
    }
    accq = wmma_f16(a, bq, accq);
    acck = wmma_f16(a, bk, acck);
    accv = wmma_f16(a, bv, accv);
  }

#pragma unroll
  for (int e = 0; e < 8; ++e) {
    int m = e + koff;
    qs[m][n0 + mrow] = accq[e];
    ks[m][n0 + mrow] = acck[e];
    vg[(r0 + m) * HEAD_DIM + n0 + mrow] = (_Float16)accv[e];
  }
  __syncthreads();

  const long bb    = r0 / SEQ_LEN;
  const long sbase = r0 - bb * SEQ_LEN;
  const float kfreq = 0.14391156831212787f;  // ln(10000)*2/128
  for (int t = tid; t < 16 * HEAD_DIM; t += 256) {
    int m = t >> 7;
    int h = t & 127;
    long sp = sbase + m;
    float inv = expf(-(float)(h & 63) * kfreq);
    float ang = (float)sp * inv;
    float c = cosf(ang), sn = sinf(ang);
    float qv = qs[m][h] * c + ((h < 64) ? -qs[m][h + 64] : qs[m][h - 64]) * sn;
    float kv = ks[m][h] * c + ((h < 64) ? -ks[m][h + 64] : ks[m][h - 64]) * sn;
    qg[(r0 + m) * HEAD_DIM + h] = (_Float16)qv;
    kTg[bb * (long)HEAD_DIM * SEQ_LEN + (long)h * SEQ_LEN + sp] = (_Float16)kv;
  }
}

// ---------------------------------------------------------------------------
// Kernel 3: causal flash attention, TDM double-buffered K/V tiles.
// ---------------------------------------------------------------------------
__global__ void __launch_bounds__(128) flash_attn_kernel(
    const _Float16* __restrict__ qg, const _Float16* __restrict__ kTg,
    const _Float16* __restrict__ vg, float* __restrict__ out) {
  __shared__ __align__(32) _Float16 kts[2][HEAD_DIM * 32];  // [buf][d][key]
  __shared__ __align__(32) _Float16 vts[2][32 * HEAD_DIM];  // [buf][key][col]
  __shared__ __align__(32) _Float16 pbuf[4][16 * 32];

  const int tid  = threadIdx.x;
  const int wave = tid >> 5;
  const int lane = tid & 31;
  const int mrow = lane & 15;
  const int koff = (lane >> 4) * 8;

  const int b  = blockIdx.x >> 6;
  const int q0 = (blockIdx.x & 63) * 64;
  const long qrow_g = (long)b * SEQ_LEN + q0 + wave * 16 + mrow;

  const _Float16* kbase = kTg + (long)b * HEAD_DIM * SEQ_LEN;
  const _Float16* vbase = vg + (long)b * SEQ_LEN * HEAD_DIM;

  // Q fragments
  v16h aq[4];
  const _Float16* qrp = qg + qrow_g * HEAD_DIM;
#pragma unroll
  for (int c = 0; c < 4; ++c) {
    v8h lo = *(const v8h*)(qrp + 32 * c + koff);
    v8h hi = *(const v8h*)(qrp + 32 * c + 16 + koff);
#pragma unroll
    for (int i = 0; i < 8; ++i) { aq[c][i] = lo[i]; aq[c][8 + i] = hi[i]; }
  }

  const float NEG_INF = -__builtin_inff();
  float mrun[8], lrun[8];
  v8f oacc[8];
  v8f zacc = {};
#pragma unroll
  for (int e = 0; e < 8; ++e) { mrun[e] = NEG_INF; lrun[e] = 0.f; }
#pragma unroll
  for (int t = 0; t < 8; ++t) oacc[t] = zacc;

  const float scale = 0.08838834764831845f;  // 1/sqrt(128)
  const int jmax = (q0 + 63) >> 5;

#if HAVE_TDM
  // prologue: TDM-load tile 0 into buffer 0 (wave 0 only; TENSORcnt tracked)
  if (wave == 0) {
    tdm_load_2d((unsigned)(size_t)&kts[0][0], kbase, 32, HEAD_DIM,
                SEQ_LEN, HEAD_DIM, SEQ_LEN);
    tdm_load_2d((unsigned)(size_t)&vts[0][0], vbase, 4096, 0, 4096, 1, 4096);
  }
#endif

  for (int j = 0; j <= jmax; ++j) {
    const int k0 = j * 32;
    const int buf = j & 1;
#if HAVE_TDM
    // issue next tile into the other buffer, then wait for this tile
    if (wave == 0) {
      if (j < jmax) {
        const int k1 = k0 + 32;
        tdm_load_2d((unsigned)(size_t)&kts[buf ^ 1][0], kbase + k1, 32,
                    HEAD_DIM, SEQ_LEN, HEAD_DIM, SEQ_LEN);
        tdm_load_2d((unsigned)(size_t)&vts[buf ^ 1][0],
                    vbase + (long)k1 * HEAD_DIM, 4096, 0, 4096, 1, 4096);
        __builtin_amdgcn_s_wait_tensorcnt(2);
      } else {
        __builtin_amdgcn_s_wait_tensorcnt(0);
      }
    }
#else
    // fallback: cooperative vector loads
    {
      const _Float16* ksrc = kbase + (long)tid * SEQ_LEN + k0;
      *(uint4*)&kts[buf][tid * 32]      = *(const uint4*)ksrc;
      *(uint4*)&kts[buf][tid * 32 + 16] = *(const uint4*)(ksrc + 16);
      const _Float16* vsrc = vbase + (long)k0 * HEAD_DIM + tid * 32;
      *(uint4*)&vts[buf][tid * 32]      = *(const uint4*)vsrc;
      *(uint4*)&vts[buf][tid * 32 + 16] = *(const uint4*)(vsrc + 16);
    }
#endif
    __syncthreads();

    const _Float16* kp = &kts[buf][0];
    const _Float16* vp = &vts[buf][0];

    // scores: two 16-key sub-tiles
    float p2[2][8];
    float tmax[8];
#pragma unroll
    for (int e = 0; e < 8; ++e) tmax[e] = NEG_INF;
#pragma unroll
    for (int t2 = 0; t2 < 2; ++t2) {
      v8f sacc = zacc;
#pragma unroll
      for (int c = 0; c < 4; ++c) {
        v16h bk;
#pragma unroll
        for (int i = 0; i < 8; ++i) {
          bk[i]     = kp[(32 * c + koff + i) * 32 + 16 * t2 + mrow];
          bk[8 + i] = kp[(32 * c + 16 + koff + i) * 32 + 16 * t2 + mrow];
        }
        sacc = wmma_f16(aq[c], bk, sacc);
      }
      const int key = k0 + 16 * t2 + mrow;
#pragma unroll
      for (int e = 0; e < 8; ++e) {
        const int qidx = q0 + wave * 16 + e + koff;
        float sv = sacc[e] * scale;
        if (key > qidx) sv = NEG_INF;
        p2[t2][e] = sv;
        tmax[e] = fmaxf(tmax[e], sv);
      }
    }
#pragma unroll
    for (int e = 0; e < 8; ++e) {
      float v = tmax[e];
      v = fmaxf(v, __shfl_xor(v, 1, 32));
      v = fmaxf(v, __shfl_xor(v, 2, 32));
      v = fmaxf(v, __shfl_xor(v, 4, 32));
      v = fmaxf(v, __shfl_xor(v, 8, 32));
      tmax[e] = v;
    }
#pragma unroll
    for (int e = 0; e < 8; ++e) {
      float newm = fmaxf(mrun[e], tmax[e]);
      float corr = expf(mrun[e] - newm);
      mrun[e] = newm;
      lrun[e] *= corr;
#pragma unroll
      for (int t = 0; t < 8; ++t) oacc[t][e] *= corr;
      float s0 = expf(p2[0][e] - newm);
      float s1 = expf(p2[1][e] - newm);
      p2[0][e] = s0;
      p2[1][e] = s1;
      float rs = s0 + s1;
      rs += __shfl_xor(rs, 1, 32);
      rs += __shfl_xor(rs, 2, 32);
      rs += __shfl_xor(rs, 4, 32);
      rs += __shfl_xor(rs, 8, 32);
      lrun[e] += rs;
    }
    // C layout -> LDS -> A fragment
#pragma unroll
    for (int t2 = 0; t2 < 2; ++t2)
#pragma unroll
      for (int e = 0; e < 8; ++e)
        pbuf[wave][(e + koff) * 32 + 16 * t2 + mrow] = (_Float16)p2[t2][e];
    __syncthreads();
    v16h ap;
    {
      v8h lo = *(const v8h*)&pbuf[wave][mrow * 32 + koff];
      v8h hi = *(const v8h*)&pbuf[wave][mrow * 32 + 16 + koff];
#pragma unroll
      for (int i = 0; i < 8; ++i) { ap[i] = lo[i]; ap[8 + i] = hi[i]; }
    }
#pragma unroll
    for (int t = 0; t < 8; ++t) {
      v16h bv;
#pragma unroll
      for (int i = 0; i < 8; ++i) {
        bv[i]     = vp[(koff + i) * HEAD_DIM + 16 * t + mrow];
        bv[8 + i] = vp[(16 + koff + i) * HEAD_DIM + 16 * t + mrow];
      }
      oacc[t] = wmma_f16(ap, bv, oacc[t]);
    }
    __syncthreads();
  }

#pragma unroll
  for (int e = 0; e < 8; ++e) {
    float inv = 1.0f / lrun[e];
    const long row = (long)b * SEQ_LEN + q0 + wave * 16 + e + koff;
#pragma unroll
    for (int t = 0; t < 8; ++t)
      out[row * HEAD_DIM + 16 * t + mrow] = oacc[t][e] * inv;
  }
}

// ---------------------------------------------------------------------------
extern "C" void kernel_launch(void* const* d_in, const int* in_sizes, int n_in,
                              void* d_out, int out_size, void* d_ws,
                              size_t ws_size, hipStream_t stream) {
  const float* x  = (const float*)d_in[0];
  const float* wq = (const float*)d_in[1];
  const float* wk = (const float*)d_in[2];
  const float* wv = (const float*)d_in[3];
  char* ws = (char*)d_ws;

  const size_t WT_BYTES  = (size_t)D_MODEL * HEAD_DIM * 2;           // 512 KB
  const size_t QKV_BYTES = (size_t)BATCH * SEQ_LEN * HEAD_DIM * 2;   // 4 MB
  _Float16* wqT = (_Float16*)(ws);
  _Float16* wkT = (_Float16*)(ws + WT_BYTES);
  _Float16* wvT = (_Float16*)(ws + 2 * WT_BYTES);
  _Float16* qg  = (_Float16*)(ws + 3 * WT_BYTES);
  _Float16* kTg = (_Float16*)(ws + 3 * WT_BYTES + QKV_BYTES);
  _Float16* vg  = (_Float16*)(ws + 3 * WT_BYTES + 2 * QKV_BYTES);
  float* out = (float*)d_out;

  dim3 gT(D_MODEL * HEAD_DIM / 256, 3);
  wtranspose_kernel<<<gT, 256, 0, stream>>>(wq, wk, wv, wqT, wkT, wvT);

  proj_rope_kernel<<<BATCH * SEQ_LEN / 16, 256, 0, stream>>>(
      x, wqT, wkT, wvT, qg, kTg, vg);

  flash_attn_kernel<<<BATCH * (SEQ_LEN / 64), 128, 0, stream>>>(
      qg, kTg, vg, out);
}
